// SwitchGRUCell_3745211482670
// MI455X (gfx1250) — compile-verified, matrix operand order checked by reading, not run on hardware
//
#include <hip/hip_runtime.h>
#include <hip/hip_bf16.h>
#include <math.h>

#define BB 128   // batch
#define DD 512   // input dim
#define HH 512   // hidden dim
#define KE 16    // experts

typedef __attribute__((ext_vector_type(16))) _Float16 v16h;
typedef __attribute__((ext_vector_type(8)))  _Float16 v8h;
typedef __attribute__((ext_vector_type(2)))  _Float16 v2h;
typedef __attribute__((ext_vector_type(8)))  float    v8f;

// ---------------------------------------------------------------------------
// Zero the 3 f32 accumulators in workspace (graph-replay safe: re-zeroed
// every launch).
// ---------------------------------------------------------------------------
__global__ void sg_zero(float* __restrict__ p, int n) {
    int i = blockIdx.x * blockDim.x + threadIdx.x;
    if (i < n) p[i] = 0.0f;
}

// ---------------------------------------------------------------------------
// Split-K expert GEMM:  acc[bank][b,h] += sum_d (u[b,k]*x[b,d]) * W[k,h,d]
// grid = (HH/128 colgroups, KE experts, 3 banks), block = 256 (8 waves)
// Each wave owns one 16-wide column strip and all 8 row tiles.
// WMMA: v_wmma_f32_16x16x32_f16, f32 accumulate.
// ---------------------------------------------------------------------------
__global__ __launch_bounds__(256)
void sg_gemm(const float* __restrict__ x,
             const float* __restrict__ u,
             const float* __restrict__ Wr,
             const float* __restrict__ Wz,
             const float* __restrict__ Wn,
             float* __restrict__ acc /* 3*BB*HH f32 */) {
    extern __shared__ _Float16 As[];          // BB x DD f16 = 128 KB LDS
    const int cg   = blockIdx.x;              // column group (128 cols)
    const int k    = blockIdx.y;              // expert
    const int bank = blockIdx.z;              // r / z / n
    const float* __restrict__ W = (bank == 0) ? Wr : ((bank == 1) ? Wz : Wn);
    float* __restrict__ accb = acc + (size_t)bank * BB * HH;

    const int tid = threadIdx.x;

    // Stage A_k = diag(u[:,k]) * x into LDS as f16, 2 elements per thread/iter.
    #pragma unroll 4
    for (int it = 0; it < (BB * DD) / (256 * 2); ++it) {
        int idx = (tid + it * 256) * 2;       // even => 4B-aligned f16 pair
        int b   = idx >> 9;                   // / DD
        float uk = u[b * KE + k];
        float2 xv = *(const float2*)(x + idx);
        v2h hv;
        hv[0] = (_Float16)(xv.x * uk);
        hv[1] = (_Float16)(xv.y * uk);
        *(v2h*)(&As[idx]) = hv;
    }
    __syncthreads();

    const int wave = tid >> 5;
    const int lane = tid & 31;
    const int n    = lane & 15;               // B column / A row / D column
    const int hs   = lane >> 4;               // half-select (K +8 offset)
    const int hcol = cg * 128 + wave * 16 + n;
    const float* __restrict__ wrow =
        W + (size_t)k * HH * DD + (size_t)hcol * DD;

    v8f c[8] = {};                            // 8 M-tiles of 16x16 f32

    for (int ch = 0; ch < DD / 32; ++ch) {
        const int dbase = ch * 32;
        // ---- B fragment: W[k, hcol, dbase + {8hs..8hs+7, 16+8hs..+7}] ----
        const float* wp = wrow + dbase + 8 * hs;
        float wbuf[16];
        *(float4*)(&wbuf[0])  = *(const float4*)(wp);
        *(float4*)(&wbuf[4])  = *(const float4*)(wp + 4);
        *(float4*)(&wbuf[8])  = *(const float4*)(wp + 16);
        *(float4*)(&wbuf[12]) = *(const float4*)(wp + 20);
        v16h bf;
        #pragma unroll
        for (int i = 0; i < 16; ++i) bf[i] = (_Float16)wbuf[i];

        // ---- 8 A fragments from LDS, one WMMA each ----
        #pragma unroll
        for (int mt = 0; mt < 8; ++mt) {
            const _Float16* ap = &As[(mt * 16 + n) * DD + dbase + 8 * hs];
            v8h a0 = *(const v8h*)(ap);        // K = 8hs .. 8hs+7
            v8h a1 = *(const v8h*)(ap + 16);   // K = 16+8hs .. 16+8hs+7
            v16h af;
            #pragma unroll
            for (int i = 0; i < 8; ++i) { af[i] = a0[i]; af[i + 8] = a1[i]; }
            c[mt] = __builtin_amdgcn_wmma_f32_16x16x32_f16(
                false, af, false, bf, (short)0, c[mt], false, false);
        }
    }

    // D layout: VGPR r -> row (r + 8*hs), column n. Split-K accumulate.
    #pragma unroll
    for (int mt = 0; mt < 8; ++mt) {
        #pragma unroll
        for (int r = 0; r < 8; ++r) {
            int bglob = mt * 16 + r + 8 * hs;
            __hip_atomic_fetch_add(&accb[(size_t)bglob * HH + hcol], c[mt][r],
                                   __ATOMIC_RELAXED, __HIP_MEMORY_SCOPE_AGENT);
        }
    }
}

// ---------------------------------------------------------------------------
// Finalize: diagonal h-path gather, bias mixes, gates.
// grid = BB blocks, block = HH threads. u[b,k] is block-uniform -> scalar.
// ---------------------------------------------------------------------------
__global__ __launch_bounds__(512)
void sg_finalize(const float* __restrict__ acc,
                 const float* __restrict__ hprev,
                 const float* __restrict__ u,
                 const float* __restrict__ Whr,
                 const float* __restrict__ Whz,
                 const float* __restrict__ Whn,
                 const float* __restrict__ bxr,
                 const float* __restrict__ bxz,
                 const float* __restrict__ bxn,
                 const float* __restrict__ brh,
                 const float* __restrict__ bzh,
                 const float* __restrict__ bnh,
                 float* __restrict__ out) {
    const int b = blockIdx.x;
    const int h = threadIdx.x;
    float sxr = 0.f, sxz = 0.f, sxn = 0.f;
    float shr = 0.f, shz = 0.f, shn = 0.f;
    float dr = 0.f, dz = 0.f, dn = 0.f;
    #pragma unroll
    for (int k = 0; k < KE; ++k) {
        float uk = u[b * KE + k];
        int kh = k * HH + h;
        sxr += uk * bxr[kh];  sxz += uk * bxz[kh];  sxn += uk * bxn[kh];
        shr += uk * brh[kh];  shz += uk * bzh[kh];  shn += uk * bnh[kh];
        size_t dix = (size_t)k * HH * HH + (size_t)h * (HH + 1);
        dr += uk * Whr[dix];  dz += uk * Whz[dix];  dn += uk * Whn[dix];
    }
    float hp = hprev[b * HH + h];
    float ar = acc[0 * BB * HH + b * HH + h] + sxr + dr * hp + shr;
    float az = acc[1 * BB * HH + b * HH + h] + sxz + dz * hp + shz;
    float r  = 1.0f / (1.0f + expf(-ar));
    float z  = 1.0f / (1.0f + expf(-az));
    float an = acc[2 * BB * HH + b * HH + h] + sxn + r * (dn * hp + shn);
    float nt = tanhf(an);
    out[b * HH + h] = (1.0f - z) * nt + z * hp;
}

// ---------------------------------------------------------------------------
extern "C" void kernel_launch(void* const* d_in, const int* in_sizes, int n_in,
                              void* d_out, int out_size, void* d_ws, size_t ws_size,
                              hipStream_t stream) {
    const float* x   = (const float*)d_in[0];
    const float* hp  = (const float*)d_in[1];
    const float* u   = (const float*)d_in[2];
    const float* Wxr = (const float*)d_in[3];
    const float* Wxz = (const float*)d_in[4];
    const float* Wxn = (const float*)d_in[5];
    const float* Whr = (const float*)d_in[6];
    const float* Whz = (const float*)d_in[7];
    const float* Whn = (const float*)d_in[8];
    const float* bxr = (const float*)d_in[9];
    const float* bxz = (const float*)d_in[10];
    const float* bxn = (const float*)d_in[11];
    const float* brh = (const float*)d_in[12];
    const float* bzh = (const float*)d_in[13];
    const float* bnh = (const float*)d_in[14];
    float* acc = (float*)d_ws;                 // 3*BB*HH f32 = 768 KB
    float* out = (float*)d_out;

    const int nacc = 3 * BB * HH;
    sg_zero<<<(nacc + 255) / 256, 256, 0, stream>>>(acc, nacc);

    dim3 grid(HH / 128, KE, 3);
    sg_gemm<<<grid, 256, BB * DD * sizeof(_Float16), stream>>>(
        x, u, Wxr, Wxz, Wxn, acc);

    sg_finalize<<<BB, HH, 0, stream>>>(acc, hp, u, Whr, Whz, Whn,
                                       bxr, bxz, bxn, brh, bzh, bnh, out);
}
